// LSTMSubTrajectoryBasedSkillGenerator_32925219291289
// MI455X (gfx1250) — matrix operand order repaired and checked by reading, not compiled, not run on hardware
//
#include <hip/hip_runtime.h>
#include <math.h>

typedef __attribute__((ext_vector_type(2)))  float  v2f;
typedef __attribute__((ext_vector_type(8)))  float  v8f;
typedef __attribute__((ext_vector_type(2)))  __bf16 v2bf;
typedef __attribute__((ext_vector_type(8)))  __bf16 v8bf;
typedef __attribute__((ext_vector_type(16))) __bf16 v16bf;

#define Bc   4096
#define Tc   128
#define OBSc 64
#define ACTc 16
#define Hc   256
#define Dc   64
#define G4H  1024
#define Kcat 320   // 64 (x) + 256 (h)

// ---- f32 WMMA: D(16x16 f32) = A(16x4 f32) * B(4x16 f32) + C ----
__device__ __forceinline__ v8f wmma4(v2f a, v2f b, v8f c) {
  return __builtin_amdgcn_wmma_f32_16x16x4_f32(false, a, false, b, (short)0, c,
                                               false, false);
}
// ---- bf16 WMMA: D(16x16 f32) = A(16x32 bf16) * B(32x16 bf16) + C ----
__device__ __forceinline__ v8f wmma32bf(v16bf a, v16bf b, v8f c) {
  return __builtin_amdgcn_wmma_f32_16x16x32_bf16(false, a, false, b, (short)0,
                                                 c, false, false);
}

__device__ __forceinline__ v16bf cat16(v8bf a, v8bf b) {
  return __builtin_shufflevector(a, b, 0, 1, 2, 3, 4, 5, 6, 7, 8, 9, 10, 11,
                                 12, 13, 14, 15);
}

__device__ __forceinline__ float sigmoidf_(float x) {
  return 1.0f / (1.0f + __expf(-x));
}

__global__ void zero_kernel(float* p, long n) {
  long i = (long)blockIdx.x * blockDim.x + threadIdx.x;
  long stride = (long)gridDim.x * blockDim.x;
  for (; i < n; i += stride) p[i] = 0.0f;
}

// ---------------------------------------------------------------------------
// Prep: split fp32 [Wx;Wh] (320 x 1024, row-major) into bf16 hi/lo arrays.
// ---------------------------------------------------------------------------
__global__ void wsplit_kernel(const float* __restrict__ Wx,
                              const float* __restrict__ Wh,
                              __bf16* __restrict__ Whi,
                              __bf16* __restrict__ Wlo) {
  long n = (long)Kcat * G4H;
  long stride = (long)gridDim.x * blockDim.x;
  for (long i = (long)blockIdx.x * blockDim.x + threadIdx.x; i < n;
       i += stride) {
    long kr = i / G4H, c = i - kr * G4H;
    float f = (kr < Dc) ? Wx[kr * G4H + c] : Wh[(kr - Dc) * G4H + c];
    __bf16 h = (__bf16)f;
    Whi[i] = h;
    Wlo[i] = (__bf16)(f - (float)h);
  }
}

// ---------------------------------------------------------------------------
// Kernel 1: x_pre[B*T, 64] = concat(obs, act) @ W_embed + b_embed  (f32 WMMA)
// ---------------------------------------------------------------------------
__global__ __launch_bounds__(256) void embed_kernel(
    const float* __restrict__ obs, const float* __restrict__ act,
    const float* __restrict__ We, const float* __restrict__ be,
    float* __restrict__ xpre) {
  __shared__ float sA[64][68];
  const int tid = threadIdx.x;
  const int lane = tid & 31;
  const int wave = tid >> 5;
  const int l = lane & 15;
  const int hi = lane >> 4;
  const int wy = wave >> 1;
  const int wx = wave & 1;
  const long mBase = (long)blockIdx.x * 64;

  v8f acc[2] = {};

  for (int i = 0; i < 16; ++i) {
    int idx = tid + i * 256;
    int r = idx >> 6, k = idx & 63;
    sA[r][k] = obs[(mBase + r) * OBSc + k];
  }
  __syncthreads();
  for (int k = 0; k < 64; k += 4) {
    int ks = k + 2 * hi;
    int m = wy * 16 + l;
    v2f a; a.x = sA[m][ks]; a.y = sA[m][ks + 1];
#pragma unroll
    for (int j = 0; j < 2; ++j) {
      int col = wx * 32 + j * 16 + l;
      v2f b; b.x = We[ks * Dc + col]; b.y = We[(ks + 1) * Dc + col];
      acc[j] = wmma4(a, b, acc[j]);
    }
  }
  __syncthreads();
  for (int i = 0; i < 4; ++i) {
    int idx = tid + i * 256;
    int r = idx >> 4, k = idx & 15;
    sA[r][k] = act[(mBase + r) * ACTc + k];
  }
  __syncthreads();
  for (int k = 0; k < 16; k += 4) {
    int ks = k + 2 * hi;
    int m = wy * 16 + l;
    v2f a; a.x = sA[m][ks]; a.y = sA[m][ks + 1];
#pragma unroll
    for (int j = 0; j < 2; ++j) {
      int col = wx * 32 + j * 16 + l;
      v2f b; b.x = We[(OBSc + ks) * Dc + col];
      b.y = We[(OBSc + ks + 1) * Dc + col];
      acc[j] = wmma4(a, b, acc[j]);
    }
  }
#pragma unroll
  for (int j = 0; j < 2; ++j) {
    int col = wx * 32 + j * 16 + l;
#pragma unroll
    for (int v = 0; v < 8; ++v) {
      long row = mBase + wy * 16 + hi * 8 + v;
      xpre[row * Dc + col] = acc[j][v] + be[col];
    }
  }
}

// ---------------------------------------------------------------------------
// Kernel 2: BN stats over B per (t,d) folded into affine (scale, bias).
// ---------------------------------------------------------------------------
__global__ __launch_bounds__(256) void bnstats_kernel(
    const float* __restrict__ xpre, const float* __restrict__ gamma,
    const float* __restrict__ beta, float* __restrict__ sBN,
    float* __restrict__ bBN) {
  __shared__ float rs[4][64];
  __shared__ float rq[4][64];
  const int t = blockIdx.x;
  const int d = threadIdx.x & 63;
  const int g = threadIdx.x >> 6;
  float s = 0.f, q = 0.f;
  for (int b = g; b < Bc; b += 4) {
    float v = xpre[(long)b * (Tc * Dc) + t * Dc + d];
    s += v; q += v * v;
  }
  rs[g][d] = s; rq[g][d] = q;
  __syncthreads();
  if (g == 0) {
    s = rs[0][d] + rs[1][d] + rs[2][d] + rs[3][d];
    q = rq[0][d] + rq[1][d] + rq[2][d] + rq[3][d];
    float mean = s * (1.0f / Bc);
    float var = q * (1.0f / Bc) - mean * mean;
    float sc = rsqrtf(var + 1e-5f) * gamma[d];
    sBN[t * Dc + d] = sc;
    bBN[t * Dc + d] = beta[d] - mean * sc;
  }
}

// ---------------------------------------------------------------------------
// Kernel 3 (x128): fused LSTM step, bf16x3 split WMMA, wide-DS fragments.
//   z = [lrelu(bn(x_t)) | h_prev] @ [Wx; Wh]
//   A*B ~= Ah*Bh + Ah*Bl + Al*Bh  (fp32 accumulate, ~2^-16 rel. error)
// LDS holds A pre-split as bf16 hi/lo (row-major, stride 48 halves = 96B so
// all fragment loads are 16B-aligned b128) and B transposed ([col][k]) so a
// lane's 16 K-values lie in two contiguous runs of 8.
// ---------------------------------------------------------------------------
__global__ __launch_bounds__(256) void lstm_step_kernel(
    const float* __restrict__ xpre, const float* __restrict__ sBN,
    const float* __restrict__ bBN, const __bf16* __restrict__ Whi,
    const __bf16* __restrict__ Wlo, const float* __restrict__ bl,
    const float* __restrict__ h_in, float* __restrict__ h_out,
    float* __restrict__ c_st, int t) {
  __shared__ __bf16 sAh[64][48];   // A hi: [row][k], K-chunk 32
  __shared__ __bf16 sAl[64][48];   // A lo
  __shared__ __bf16 sBh[128][48];  // B hi transposed: [gatecol][k]
  __shared__ __bf16 sBl[128][48];  // B lo transposed
  const int tid = threadIdx.x;
  const int lane = tid & 31;
  const int wave = tid >> 5;
  const int l = lane & 15;
  const int hi16 = lane >> 4;
  const int wy = wave >> 1;
  const int wx = wave & 1;
  const long rowBase = (long)blockIdx.x * 64;
  const int hBase = blockIdx.y * 32;

  v8f acc[4] = {};  // gates i,f,g,o

  for (int kc = 0; kc < 10; ++kc) {  // K = 320 in chunks of 32
    // ---- stage A chunk: 64 rows x 32 K, two K per thread, split hi/lo ----
    for (int i = 0; i < 4; ++i) {
      int idx = tid + i * 256;
      int r = idx >> 4;
      int k = (idx & 15) * 2;
      float f0, f1;
      if (kc < 2) {  // x part: K 0..63 (chunk boundary == split boundary)
        int kabs = kc * 32 + k;
        long base = (rowBase + r) * (long)(Tc * Dc) + t * Dc + kabs;
        f0 = xpre[base];
        f1 = xpre[base + 1];
        f0 = f0 * sBN[t * Dc + kabs] + bBN[t * Dc + kabs];
        f1 = f1 * sBN[t * Dc + kabs + 1] + bBN[t * Dc + kabs + 1];
        f0 = (f0 >= 0.f) ? f0 : 0.2f * f0;
        f1 = (f1 >= 0.f) ? f1 : 0.2f * f1;
      } else {       // h part
        long base = (rowBase + r) * (long)Hc + (kc * 32 + k - Dc);
        f0 = h_in[base];
        f1 = h_in[base + 1];
      }
      __bf16 h0 = (__bf16)f0, h1 = (__bf16)f1;
      v2bf vh = {h0, h1};
      v2bf vl = {(__bf16)(f0 - (float)h0), (__bf16)(f1 - (float)h1)};
      *(v2bf*)&sAh[r][k] = vh;
      *(v2bf*)&sAl[r][k] = vl;
    }
    // ---- stage B chunk transposed: sB[g*32+n][kr] = W[kabs][g*256+hB+n] ---
    for (int i = 0; i < 8; ++i) {
      int idx = tid + i * 256;
      int col = idx & 127;
      int kr = (idx >> 7) * 2;
      int gg = col >> 5, n = col & 31;
      long w0 = (long)(kc * 32 + kr) * G4H + gg * Hc + hBase + n;
      v2bf vh = {Whi[w0], Whi[w0 + G4H]};
      v2bf vl = {Wlo[w0], Wlo[w0 + G4H]};
      *(v2bf*)&sBh[col][kr] = vh;
      *(v2bf*)&sBl[col][kr] = vl;
    }
    __syncthreads();

    // ---- A fragments: two aligned b128 loads per hi/lo + concat ----
    // 16-bit A 16x32: lane(<16) K = {0..7} U {16..23}; lanes 16-31: +8.
    const int m = wy * 16 + l;
    const int aoff = hi16 * 8;
    v16bf ah = cat16(*(const v8bf*)&sAh[m][aoff],
                     *(const v8bf*)&sAh[m][aoff + 16]);
    v16bf al = cat16(*(const v8bf*)&sAl[m][aoff],
                     *(const v8bf*)&sAl[m][aoff + 16]);
    // ---- B fragments: K = 8*laneHi + {0..7} U {16..23}, same as A ----
    const int boff = hi16 * 8;
#pragma unroll
    for (int gg = 0; gg < 4; ++gg) {
      int col = gg * 32 + wx * 16 + l;
      v16bf bh = cat16(*(const v8bf*)&sBh[col][boff],
                       *(const v8bf*)&sBh[col][boff + 16]);
      v16bf blo = cat16(*(const v8bf*)&sBl[col][boff],
                        *(const v8bf*)&sBl[col][boff + 16]);
      acc[gg] = wmma32bf(ah, bh, acc[gg]);
      acc[gg] = wmma32bf(ah, blo, acc[gg]);
      acc[gg] = wmma32bf(al, bh, acc[gg]);
    }
    __syncthreads();
  }

  // ---- gate math: flax order i,f,g,o; bias added once per step ----
  const int hcol = hBase + wx * 16 + l;
  const float bi = bl[0 * Hc + hcol];
  const float bf_ = bl[1 * Hc + hcol];
  const float bg = bl[2 * Hc + hcol];
  const float bo = bl[3 * Hc + hcol];
#pragma unroll
  for (int v = 0; v < 8; ++v) {
    long row = rowBase + wy * 16 + hi16 * 8 + v;
    float zi = acc[0][v] + bi;
    float zf = acc[1][v] + bf_;
    float zg = acc[2][v] + bg;
    float zo = acc[3][v] + bo;
    float cp = c_st[row * Hc + hcol];
    float cn = sigmoidf_(zf) * cp + sigmoidf_(zi) * tanhf(zg);
    float hn = sigmoidf_(zo) * tanhf(cn);
    c_st[row * Hc + hcol] = cn;
    h_out[row * Hc + hcol] = hn;
  }
}

// ---------------------------------------------------------------------------
// Kernel 4: mu = h@Wmu + bmu ; log_std = clip(h@Wls + bls, -10, 2)  (f32 WMMA)
// ---------------------------------------------------------------------------
__global__ __launch_bounds__(256) void heads_kernel(
    const float* __restrict__ h, const float* __restrict__ Wmu,
    const float* __restrict__ bmu, const float* __restrict__ Wls,
    const float* __restrict__ bls, float* __restrict__ out) {
  __shared__ float sA[64][68];
  const int tid = threadIdx.x;
  const int lane = tid & 31;
  const int wave = tid >> 5;
  const int l = lane & 15;
  const int hi = lane >> 4;
  const int wy = wave >> 1;
  const int wx = wave & 1;
  const long mBase = (long)blockIdx.x * 64;

  v8f acc[4] = {};

  for (int kc = 0; kc < 4; ++kc) {
    for (int i = 0; i < 16; ++i) {
      int idx = tid + i * 256;
      int r = idx >> 6, k = idx & 63;
      sA[r][k] = h[(mBase + r) * Hc + kc * 64 + k];
    }
    __syncthreads();
    for (int k = 0; k < 64; k += 4) {
      int ks = k + 2 * hi;
      int m = wy * 16 + l;
      v2f a; a.x = sA[m][ks]; a.y = sA[m][ks + 1];
      int kr = kc * 64 + ks;
#pragma unroll
      for (int j = 0; j < 2; ++j) {
        int col = wx * 32 + j * 16 + l;
        v2f bm; bm.x = Wmu[kr * Dc + col]; bm.y = Wmu[(kr + 1) * Dc + col];
        acc[j] = wmma4(a, bm, acc[j]);
        v2f bv; bv.x = Wls[kr * Dc + col]; bv.y = Wls[(kr + 1) * Dc + col];
        acc[2 + j] = wmma4(a, bv, acc[2 + j]);
      }
    }
    __syncthreads();
  }
#pragma unroll
  for (int j = 0; j < 2; ++j) {
    int col = wx * 32 + j * 16 + l;
#pragma unroll
    for (int v = 0; v < 8; ++v) {
      long row = mBase + wy * 16 + hi * 8 + v;
      out[row * Dc + col] = acc[j][v] + bmu[col];
      float ls = acc[2 + j][v] + bls[col];
      ls = fminf(fmaxf(ls, -10.0f), 2.0f);
      out[(long)Bc * Dc + row * Dc + col] = ls;
    }
  }
}

extern "C" void kernel_launch(void* const* d_in, const int* in_sizes, int n_in,
                              void* d_out, int out_size, void* d_ws,
                              size_t ws_size, hipStream_t stream) {
  const float* obs = (const float*)d_in[0];
  const float* act = (const float*)d_in[1];
  const float* We  = (const float*)d_in[2];
  const float* be  = (const float*)d_in[3];
  const float* gam = (const float*)d_in[4];
  const float* bet = (const float*)d_in[5];
  const float* Wx  = (const float*)d_in[6];
  const float* Wh  = (const float*)d_in[7];
  const float* bl  = (const float*)d_in[8];
  const float* Wmu = (const float*)d_in[9];
  const float* bmu = (const float*)d_in[10];
  const float* Wls = (const float*)d_in[11];
  const float* bls = (const float*)d_in[12];
  float* out = (float*)d_out;

  float* ws   = (float*)d_ws;
  float* xpre = ws;                          // B*T*D  (128 MB)
  float* sBN  = xpre + (long)Bc * Tc * Dc;   // T*D
  float* bBN  = sBN + Tc * Dc;               // T*D
  float* h0   = bBN + Tc * Dc;               // B*H
  float* h1   = h0 + (long)Bc * Hc;          // B*H
  float* cst  = h1 + (long)Bc * Hc;          // B*H
  __bf16* Whi = (__bf16*)(cst + (long)Bc * Hc);  // 320*1024 bf16
  __bf16* Wlo = Whi + (long)Kcat * G4H;          // 320*1024 bf16

  zero_kernel<<<512, 256, 0, stream>>>(h0, (long)Bc * Hc);
  zero_kernel<<<512, 256, 0, stream>>>(cst, (long)Bc * Hc);
  wsplit_kernel<<<320, 256, 0, stream>>>(Wx, Wh, Whi, Wlo);

  embed_kernel<<<(Bc * Tc) / 64, 256, 0, stream>>>(obs, act, We, be, xpre);
  bnstats_kernel<<<Tc, 256, 0, stream>>>(xpre, gam, bet, sBN, bBN);

  for (int t = 0; t < Tc; ++t) {
    float* hin  = (t & 1) ? h1 : h0;
    float* hout = (t & 1) ? h0 : h1;
    lstm_step_kernel<<<dim3(Bc / 64, Hc / 32), 256, 0, stream>>>(
        xpre, sBN, bBN, Whi, Wlo, bl, hin, hout, cst, t);
  }
  // T=128: last step (t=127, odd) writes h0.
  heads_kernel<<<Bc / 64, 256, 0, stream>>>(h0, Wmu, bmu, Wls, bls, out);
}
